// TriAttention_9302899163644
// MI455X (gfx1250) — matrix operand order repaired and checked by reading, not compiled, not run on hardware
//
#include <hip/hip_runtime.h>

// ---------------------------------------------------------------------------
// TriAttention for MI455X (gfx1250, wave32, WMMA).
// All GEMMs: v_wmma_f32_16x16x32_bf16; one wave computes a 32(M) x 64(N) tile
// (2 A-frags x 4 B-frags -> 8 accumulator chains). The k-loop is branchless:
// invalid B columns are handled by clamping load addresses and zeroing in the
// epilogue, so no exec-mask churn or accumulator copies around the WMMAs.
// Pipeline:
//   0) bf16 conversions; memory -> memT[b][h][s]; W_tri -> wtriT[(j,r)][i][k]
//   1) three 2-layer MLPs -> X=[head,1,pad416], Y=[mid,1,pad416], Z=tail
//   2) per (b, 32-j chunk):
//        GEMM-A: ZW[z][c][i] = sum_k Z[z][k] * wtriT[jbase*4+c][i][k]
//        GEMM-B: T2[x][r][z][jl] = sum_i X[x][i] * ZW[z][c=jl*4+r][i] (perm-C)
//        GEMM-C: score[b][x][r][z][y] += sum_jl T2[x][r][z][jl]*Y[y][jbase+jl]
//   3) mask + softmax over z -> alpha[b][r][x][y][z] (bf16)
//   4) fused: relu(alpha @ memT^T) . Vw[r,:] + Vb -> out[b][x][y][r]
// ---------------------------------------------------------------------------

typedef __attribute__((ext_vector_type(16))) __bf16 v16bf;
typedef __attribute__((ext_vector_type(8)))  float  v8f;

union Frag { v16bf v; unsigned short s[16]; uint4 q[2]; };

#define FLAG_RELU  1
#define FLAG_ACCUM 2
#define FLAG_BF16  4
#define FLAG_PERMC 8   // store col (nd,nm) at (nm&3)*4096 + nd*32 + (nm>>2)

__device__ __forceinline__ unsigned short f2bf(float f) {
    unsigned int u = __float_as_uint(f);
    unsigned int r = u + 0x7FFFu + ((u >> 16) & 1u);   // round-to-nearest-even
    return (unsigned short)(r >> 16);
}

// ---------------------------------------------------------------------------
__global__ void cvt_f32_bf16(const float* __restrict__ s,
                             unsigned short* __restrict__ d, long n) {
    long i = (long)blockIdx.x * blockDim.x + threadIdx.x;
    long stride = (long)gridDim.x * blockDim.x;
    for (; i < n; i += stride) d[i] = f2bf(s[i]);
}

// memory[b][s][h] (f32) -> memT[b][h][s] (bf16)
__global__ void transpose_bsh(const float* __restrict__ m,
                              unsigned short* __restrict__ mt) {
    long i = (long)blockIdx.x * blockDim.x + threadIdx.x;
    if (i >= 2L * 128 * 768) return;
    int h = (int)(i % 768);
    long t = i / 768;
    int s = (int)(t % 128);
    int b = (int)(t / 128);
    mt[((long)b * 768 + h) * 128 + s] = f2bf(m[i]);
}

// W_tri[i][k][jr] (f32, jr = j*4+r, 1540 wide) -> wtriT[jr][i][k] (bf16)
// LDS-tiled 32x32 transpose: coalesced f32 reads over jr, k-contiguous writes.
__global__ void transpose_wtri(const float* __restrict__ W,
                               unsigned short* __restrict__ WT) {
    const int jt = blockIdx.x;        // 0..48  (jr tile)
    const int kt = blockIdx.y;        // 0..11  (k tile)
    const int i  = blockIdx.z;        // 0..384
    const int jr0 = jt * 32, k0 = kt * 32;
    __shared__ float t[32][33];

    for (int kk = threadIdx.y; kk < 32; kk += 8) {
        int jr = jr0 + threadIdx.x;
        float v = 0.f;
        if (jr < 1540) v = W[(long)i * 591360 + (long)(k0 + kk) * 1540 + jr];
        t[kk][threadIdx.x] = v;
    }
    __syncthreads();
    for (int jj = threadIdx.y; jj < 32; jj += 8) {
        int jr = jr0 + jj;
        if (jr < 1540)
            WT[((long)jr * 385 + i) * 384 + k0 + threadIdx.x] =
                f2bf(t[threadIdx.x][jj]);
    }
}

// X/Y pad columns: col 384 = 1.0 (bias feature), cols 385..415 = 0
__global__ void init_pad(unsigned short* __restrict__ X,
                         unsigned short* __restrict__ Y) {
    int row = blockIdx.x;
    int c   = threadIdx.x;
    unsigned short v = (c == 0) ? (unsigned short)0x3F80u : (unsigned short)0;
    X[(long)row * 416 + 384 + c] = v;
    Y[(long)row * 416 + 384 + c] = v;
}

// ---------------------------------------------------------------------------
// bf16 WMMA GEMM: one wave computes 32(M) x 64(N); K multiple of 32; all
// operands K-contiguous (b128 loads only). Branchless inner loop.
//   A element (m,k) at A[(m/aMod)*aHi + (m%aMod)*aLo + k]
//   B element (n,k) at B[(n/bMod)*bHi + (n%bMod)*bLo + bExtra + k]
//   columns with n/bMod >= bDivLimit or n%bMod >= bModLimit produce zeros
//   (loads are clamped to a valid column; the epilogue writes 0 instead).
__global__ void gemm_bf16_wmma(
    const unsigned short* __restrict__ A,
    const unsigned short* __restrict__ B,
    const float* __restrict__ bias,
    float* __restrict__ Cf,
    unsigned short* __restrict__ Cb,
    int M, int N, int K,
    int aMod, long aHi, int aLo,
    int bMod, long bHi, int bLo, long bExtra,
    int bDivLimit, int bModLimit,
    long ldc, int flags)
{
    const int lane = threadIdx.x;            // 0..31
    const int tN = blockIdx.x * 64;
    const int tM = blockIdx.y * 32;

    const int kha = (lane >> 4) * 8;         // A: K segs {kha..+7, kha+16..+23}
    const int khb = (lane >> 4) * 8;         // B: K segs {khb..+7, khb+16..+23}

    long aB[2];
#pragma unroll
    for (int h = 0; h < 2; ++h) {
        const int am = tM + h * 16 + (lane & 15);
        aB[h] = (long)(am / aMod) * aHi + (long)(am % aMod) * aLo;
    }

    long bB[4]; bool bv[4]; int bnd[4], bnm[4];
#pragma unroll
    for (int s = 0; s < 4; ++s) {
        const int bn = tN + s * 16 + (lane & 15);
        bnd[s] = bn / bMod;
        bnm[s] = bn % bMod;
        bv[s]  = (bnd[s] < bDivLimit) && (bnm[s] < bModLimit);
        // clamp load address to a valid column; result zeroed in epilogue
        const int ndc = bnd[s] < bDivLimit ? bnd[s] : (bDivLimit - 1);
        const int nmc = bnm[s] < bModLimit ? bnm[s] : (bModLimit - 1);
        bB[s] = (long)ndc * bHi + (long)nmc * bLo + bExtra;
    }

    v8f acc[4][2];
#pragma unroll
    for (int s = 0; s < 4; ++s)
#pragma unroll
        for (int h = 0; h < 2; ++h)
            acc[s][h] = (v8f){0.f,0.f,0.f,0.f,0.f,0.f,0.f,0.f};

    for (int k0 = 0; k0 < K; k0 += 32) {
        Frag a0, a1;
        {
            const unsigned short* pa0 = A + aB[0] + k0 + kha;
            const unsigned short* pa1 = A + aB[1] + k0 + kha;
            a0.q[0] = *(const uint4*)(pa0);
            a0.q[1] = *(const uint4*)(pa0 + 16);
            a1.q[0] = *(const uint4*)(pa1);
            a1.q[1] = *(const uint4*)(pa1 + 16);
        }
#pragma unroll
        for (int s = 0; s < 4; ++s) {
            Frag b;
            const unsigned short* pb = B + bB[s] + k0 + khb;
            b.q[0] = *(const uint4*)(pb);
            b.q[1] = *(const uint4*)(pb + 16);
            if (s == 0)
                __builtin_prefetch((const void*)(pb + 64), 0, 1);
            acc[s][0] = __builtin_amdgcn_wmma_f32_16x16x32_bf16(
                false, a0.v, false, b.v, (short)0, acc[s][0], false, false);
            acc[s][1] = __builtin_amdgcn_wmma_f32_16x16x32_bf16(
                false, a1.v, false, b.v, (short)0, acc[s][1], false, false);
        }
    }

    const int rowOff = (lane >> 4) * 8;      // lanes 16-31 own rows M+8..M+15
#pragma unroll
    for (int s = 0; s < 4; ++s) {
        const int cn = tN + s * 16 + (lane & 15);
        const float bvadd = (bias && bv[s]) ? bias[cn] : 0.f;
        long col;
        if (flags & FLAG_PERMC)
            col = (long)(bnm[s] & 3) * 4096 + (long)bnd[s] * 32 + (bnm[s] >> 2);
        else
            col = cn;
#pragma unroll
        for (int h = 0; h < 2; ++h) {
#pragma unroll
            for (int p = 0; p < 8; ++p) {
                const int cm = tM + h * 16 + rowOff + p;
                float v = acc[s][h][p] + bvadd;
                if (flags & FLAG_RELU) v = v > 0.f ? v : 0.f;
                if (!bv[s]) v = 0.f;         // invalid column -> zero output
                const long ci = (long)cm * ldc + col;
                if (flags & FLAG_ACCUM) {
                    if (bv[s]) Cf[ci] += v;
                } else if (flags & FLAG_BF16) {
                    Cb[ci] = f2bf(v);
                } else {
                    Cf[ci] = v;
                }
            }
        }
    }
}

// ---------------------------------------------------------------------------
// mask + softmax over z.
// score layout [b][x][r][z][y] (f32), alpha layout [b][r][x][y][z] (bf16)
__global__ void mask_softmax(const float* __restrict__ score,
                             unsigned short* __restrict__ alpha) {
    const int bid = blockIdx.x;              // (b,x,r): b*512 + x*4 + r
    const int r = bid & 3;
    const int x = (bid >> 2) & 127;
    const int b = bid >> 9;
    const int y = threadIdx.x;               // 0..127
    const float NEG = -1000000.0f;

    const float* sp = score + (((long)(b * 128 + x) * 4 + r) * 128) * 128 + y;
    float mx = -3.4e38f;
    for (int z = 0; z < 128; ++z) {
        float v = sp[(long)z * 128];
        if (z > y || z < x) v = NEG;
        mx = fmaxf(mx, v);
    }
    float sum = 0.f;
    for (int z = 0; z < 128; ++z) {
        float v = sp[(long)z * 128];
        if (z > y || z < x) v = NEG;
        sum += __expf(v - mx);
    }
    const float inv = 1.f / sum;
    unsigned short* ap = alpha + ((((long)(b * 4 + r) * 128 + x) * 128) + y) * 128;
    for (int z = 0; z < 128; ++z) {
        float v = sp[(long)z * 128];
        if (z > y || z < x) v = NEG;
        ap[z] = f2bf(__expf(v - mx) * inv);
    }
}

// ---------------------------------------------------------------------------
// Fused final stage: out[b][x][y][r] =
//   sum_h relu( sum_z alpha[b][r][x][y][z] * memT[b][h][z] ) * Vw[r][h] + Vb[r]
__global__ void attn_out_fused(const unsigned short* __restrict__ alpha,
                               const unsigned short* __restrict__ memT,
                               const float* __restrict__ Vw,
                               const float* __restrict__ Vb,
                               float* __restrict__ out) {
    const int bid = blockIdx.x;              // 8192 = b(2)*r(4)*x(128)*yt(8)
    const int yt = bid & 7;
    const int x  = (bid >> 3) & 127;
    const int r  = (bid >> 10) & 3;
    const int b  = bid >> 12;
    const int lane = threadIdx.x;

    const long arow = ((((long)(b * 4 + r) * 128 + x) * 128) + yt * 16 + (lane & 15)) * 128;
    const int kha = (lane >> 4) * 8;
    Frag a[4];
#pragma unroll
    for (int s = 0; s < 4; ++s) {
        const unsigned short* pa = alpha + arow + s * 32 + kha;
        a[s].q[0] = *(const uint4*)(pa);
        a[s].q[1] = *(const uint4*)(pa + 16);
    }

    const unsigned short* mb = memT + (long)b * 768 * 128;
    const int hl  = lane & 15;
    const int khb = (lane >> 4) * 8;
    float pr[8] = {0.f, 0.f, 0.f, 0.f, 0.f, 0.f, 0.f, 0.f};

    for (int ht = 0; ht < 48; ++ht) {
        const int h = ht * 16 + hl;
        v8f acc = {0.f, 0.f, 0.f, 0.f, 0.f, 0.f, 0.f, 0.f};
#pragma unroll
        for (int s = 0; s < 4; ++s) {
            Frag bf;
            const unsigned short* pb = mb + (long)h * 128 + s * 32 + khb;
            bf.q[0] = *(const uint4*)(pb);
            bf.q[1] = *(const uint4*)(pb + 16);
            acc = __builtin_amdgcn_wmma_f32_16x16x32_bf16(
                false, a[s].v, false, bf.v, (short)0, acc, false, false);
        }
        const float w = Vw[r * 768 + h];
#pragma unroll
        for (int p = 0; p < 8; ++p) {
            float v = acc[p];
            v = v > 0.f ? v : 0.f;           // relu(type_span_h)
            pr[p] += v * w;
        }
    }
#pragma unroll
    for (int p = 0; p < 8; ++p)
        for (int m = 1; m < 16; m <<= 1)
            pr[p] += __shfl_xor(pr[p], m, 16);

    if ((lane & 15) == 0) {
        const int ybase = yt * 16 + (lane >> 4) * 8;
        const float vb = Vb[r];
#pragma unroll
        for (int p = 0; p < 8; ++p) {
            const int y = ybase + p;
            out[(((long)b * 128 + x) * 128 + y) * 4 + r] = pr[p] + vb;
        }
    }
}

// ---------------------------------------------------------------------------
extern "C" void kernel_launch(void* const* d_in, const int* in_sizes, int n_in,
                              void* d_out, int out_size, void* d_ws, size_t ws_size,
                              hipStream_t stream) {
    const float* memory = (const float*)d_in[0];
    const float* Wh1 = (const float*)d_in[1];  const float* bh1 = (const float*)d_in[2];
    const float* Wh2 = (const float*)d_in[3];  const float* bh2 = (const float*)d_in[4];
    const float* Wt1 = (const float*)d_in[5];  const float* bt1 = (const float*)d_in[6];
    const float* Wt2 = (const float*)d_in[7];  const float* bt2 = (const float*)d_in[8];
    const float* Wm1 = (const float*)d_in[9];  const float* bm1 = (const float*)d_in[10];
    const float* Wm2 = (const float*)d_in[11]; const float* bm2 = (const float*)d_in[12];
    const float* Wtri = (const float*)d_in[13];
    const float* Vw  = (const float*)d_in[14]; const float* Vb  = (const float*)d_in[15];
    float* out = (float*)d_out;

    char* wp = (char*)d_ws;
    auto carve = [&](size_t bytes) {
        char* p = wp;
        wp += (bytes + 255) & ~(size_t)255;
        return p;
    };
    const long NTRI = 227673600L;                       // 1540*385*384
    unsigned short* wtriT = (unsigned short*)carve((size_t)NTRI * 2);
    unsigned short* memB  = (unsigned short*)carve(196608 * 2);
    unsigned short* memT  = (unsigned short*)carve(196608 * 2);
    unsigned short* W1b[3], *W2b[3];
    for (int i = 0; i < 3; ++i) W1b[i] = (unsigned short*)carve(294912 * 2);
    for (int i = 0; i < 3; ++i) W2b[i] = (unsigned short*)carve(147456 * 2);
    unsigned short* h1 = (unsigned short*)carve(98304 * 2);       // [256][384]
    unsigned short* Xb = (unsigned short*)carve(106496 * 2);      // [256][416]
    unsigned short* Yb = (unsigned short*)carve(106496 * 2);      // [256][416]
    unsigned short* Zb = (unsigned short*)carve(98304 * 2);       // [256][384]
    unsigned short* ZW = (unsigned short*)carve(6815744L * 2);    // [128][128][416]
    unsigned short* T2 = (unsigned short*)carve(2097152L * 2);    // [128][4][128][32]
    float*          score = (float*)carve(16777216L * 4);         // [2][128][4][128][128]
    unsigned short* alpha = (unsigned short*)carve(16777216L * 2);// [2][4][128][128][128]

    const dim3 blk(32);
    const int BIG = 1 << 30;

    // ---- stage 0: conversions / transposes -------------------------------
    transpose_wtri<<<dim3(49, 12, 385), dim3(32, 8), 0, stream>>>(Wtri, wtriT);
    cvt_f32_bf16<<<256, 256, 0, stream>>>(memory, memB, 196608);
    cvt_f32_bf16<<<512, 256, 0, stream>>>(Wh1, W1b[0], 294912);
    cvt_f32_bf16<<<512, 256, 0, stream>>>(Wt1, W1b[1], 294912);
    cvt_f32_bf16<<<512, 256, 0, stream>>>(Wm1, W1b[2], 294912);
    cvt_f32_bf16<<<512, 256, 0, stream>>>(Wh2, W2b[0], 147456);
    cvt_f32_bf16<<<512, 256, 0, stream>>>(Wt2, W2b[1], 147456);
    cvt_f32_bf16<<<512, 256, 0, stream>>>(Wm2, W2b[2], 147456);
    transpose_bsh<<<(196608 + 255) / 256, 256, 0, stream>>>(memory, memT);
    init_pad<<<256, 32, 0, stream>>>(Xb, Yb);

    // ---- stage 1: three 2-layer MLPs -------------------------------------
    const float* bias1[3] = {bh1, bt1, bm1};
    const float* bias2[3] = {bh2, bt2, bm2};
    unsigned short* dst2[3] = {Xb, Zb, Yb};   // head->X, tail->Z, mid->Y
    long ldc2[3] = {416, 384, 416};
    for (int m = 0; m < 3; ++m) {
        gemm_bf16_wmma<<<dim3(384 / 64, 256 / 32), blk, 0, stream>>>(
            memB, W1b[m], bias1[m], nullptr, h1,
            256, 384, 768,
            1, 768L, 0,
            1, 768L, 0, 0L,
            BIG, 1,
            384L, FLAG_RELU | FLAG_BF16);
        gemm_bf16_wmma<<<dim3(384 / 64, 256 / 32), blk, 0, stream>>>(
            h1, W2b[m], bias2[m], nullptr, dst2[m],
            256, 384, 384,
            1, 384L, 0,
            1, 384L, 0, 0L,
            BIG, 1,
            ldc2[m], FLAG_RELU | FLAG_BF16);
    }

    // ---- stage 2: tri-affine score, chunked over 32 j-values -------------
    // wtriT elem offset = (j*4+r)*147840 + i*384 + k
    for (int b = 0; b < 2; ++b) {
        const unsigned short* Zp = Zb + (long)b * 128 * 384;
        const unsigned short* Xp = Xb + (long)b * 128 * 416;
        const unsigned short* Yp = Yb + (long)b * 128 * 416;
        float* scoreB = score + (long)b * 128 * 4 * 128 * 128;
        for (int chunk = 0; chunk < 13; ++chunk) {
            const int jbase = chunk * 32;
            int cLim = (385 - jbase) * 4;            // valid c = jloc*4+r count
            if (cLim > 128) cLim = 128;
            if (cLim < 4) cLim = 4;

            // GEMM-A: ZW[z][c][i] = sum_k Z[z][k] * wtriT[jbase*4+c][i][k]
            //   n = c*416 + i  -> nd=c (stride 147840), nm=i (stride 384)
            gemm_bf16_wmma<<<dim3(53248 / 64, 128 / 32), blk, 0, stream>>>(
                Zp, wtriT, nullptr, nullptr, ZW,
                128, 53248, 384,
                1, 384L, 0,
                416, 147840L, 384, (long)jbase * 4 * 147840,
                cLim, 385,
                53248L, FLAG_BF16);

            // GEMM-B: T2[x][r][z][jl] = sum_i X[x][i] * ZW[z][c][i]
            //   n = z*128 + c -> nd=z (stride 53248), nm=c (stride 416)
            //   PERMC: col -> (c&3)*4096 + z*32 + (c>>2)
            gemm_bf16_wmma<<<dim3(16384 / 64, 128 / 32), blk, 0, stream>>>(
                Xp, ZW, nullptr, nullptr, T2,
                128, 16384, 416,
                1, 416L, 0,
                128, 53248L, 416, 0L,
                BIG, BIG,
                16384L, FLAG_BF16 | FLAG_PERMC);

            // GEMM-C: score[(x,r,z)][y] += sum_jl T2[x][r][z][jl]*Y[y][jbase+jl]
            //   A: m=(x*4+r)*128+z -> m/128 stride 4096, m%128 stride 32
            gemm_bf16_wmma<<<dim3(128 / 64, 65536 / 32), blk, 0, stream>>>(
                T2, Yp, nullptr, scoreB, nullptr,
                65536, 128, 32,
                128, 4096L, 32,
                1, 416L, 0, (long)jbase,
                BIG, 1,
                128L, (chunk == 0) ? 0 : FLAG_ACCUM);
        }
    }

    // ---- stage 3: mask + softmax over z ----------------------------------
    mask_softmax<<<1024, 128, 0, stream>>>(score, alpha);

    // ---- stage 4: fused alpha @ memory -> relu -> .Vw + Vb ---------------
    attn_out_fused<<<8192, 32, 0, stream>>>(alpha, memT, Vw, Vb, out);
}
